// OnePieceTransformer_59322088292926
// MI455X (gfx1250) — matrix operand order repaired and checked by reading, not compiled, not run on hardware
//
#include <hip/hip_runtime.h>
#include <hip/hip_bf16.h>
#include <cmath>

#define VOCAB 1500
#define NE    208
#define TT    256      // block_size (sequence length)
#define NHD   4        // n_head
#define HD    52       // head_size
#define NL    8        // n_layer
#define BB    256      // batch
#define BT    (BB*TT)  // 65536 token rows
#define FF    (4*NE)   // 832

typedef __attribute__((ext_vector_type(16))) __bf16 v16bf;
typedef __attribute__((ext_vector_type(8)))  float  v8f;
typedef __attribute__((ext_vector_type(4)))  float  f32x4;
typedef unsigned short u16;
typedef __attribute__((ext_vector_type(4)))  u16 u16x4;
typedef __attribute__((ext_vector_type(8)))  u16 u16x8;
typedef __attribute__((ext_vector_type(16))) u16 u16x16;

__device__ __forceinline__ u16 f2bf(float f) {
  unsigned int u = __float_as_uint(f);
  u += 0x7FFFu + ((u >> 16) & 1u);      // round-to-nearest-even
  return (u16)(u >> 16);
}

// ---- WMMA fragment loaders (ISA 7.12.2 layouts, wave32) -------------------
// A (16x32 bf16): lane L -> row = L&15 ; kbase = (L>>4)*8
//   elems j=0..7 -> K = kbase+j ; j=8..15 -> K = kbase+16+(j-8)
__device__ __forceinline__ v16bf load_frag_A(const u16* rowp, int kbase) {
  u16x8 lo = *(const u16x8*)(rowp + kbase);
  u16x8 hi = *(const u16x8*)(rowp + kbase + 16);
  u16x16 r;
#pragma unroll
  for (int i = 0; i < 8; ++i) { r[i] = lo[i]; r[i + 8] = hi[i]; }
  return __builtin_bit_cast(v16bf, r);
}
// B (32x16 bf16), staged N-major in LDS (Bs[n][k]): lane L -> col = L&15 ;
//   kb = (L>>4)*8 ; elems j=0..7 -> K = kb+j ; j=8..15 -> K = kb+16+(j-8)
__device__ __forceinline__ v16bf load_frag_B(const u16* colp, int kb) {
  u16x8 lo = *(const u16x8*)(colp + kb);
  u16x8 hi = *(const u16x8*)(colp + kb + 16);
  u16x16 r;
#pragma unroll
  for (int i = 0; i < 8; ++i) { r[i] = lo[i]; r[i + 8] = hi[i]; }
  return __builtin_bit_cast(v16bf, r);
}

__device__ __forceinline__ v8f wmma_bf16(v16bf a, v16bf b, v8f c) {
  return __builtin_amdgcn_wmma_f32_16x16x32_bf16(
      /*neg_a=*/false, a, /*neg_b=*/false, b,
      /*c_mod=*/(short)0, c, /*reuse_a=*/false, /*reuse_b=*/false);
}

// ---- Generic batched bf16 WMMA GEMM ---------------------------------------
// A: bf16 row-major. B: transB ? bf16 N-major : f32 K-major (V only).
// Cf (f32) and/or Cbf (bf16) outputs share indexing; bias/relu/resid on the
// f32 accumulator. All M multiples of 128; all N,K multiples of 4.
#define BM 128
#define BN 128
#define BK 64
#define LDSK 72   // 64 + 8 u16 pad: 144B row stride (16B-aligned)

__global__ __launch_bounds__(256) void gemm_bf16_kernel(
    const u16* __restrict__ A, const void* __restrict__ Bv,
    const float* __restrict__ bias, const float* __restrict__ resid,
    float* __restrict__ Cf, u16* __restrict__ Cbf,
    int M, int N, int K, int lda, int ldb, int ldc,
    long long sAb, long long sAh, long long sBb, long long sBh,
    long long sCb, long long sCh, int nH, int transB, int relu)
{
  __shared__ __align__(16) u16 As[BM][LDSK];
  __shared__ __align__(16) u16 Bs[BN][LDSK];

  const int tid  = threadIdx.x;
  const int lane = tid & 31;
  const int wave = tid >> 5;
  const int wm   = (wave >> 1) * 32;   // wave row in block tile: 0,32,64,96
  const int wn   = (wave & 1) * 64;    // wave col in block tile: 0,64

  const int z  = blockIdx.z;
  const int zb = z / nH, zh = z % nH;
  const long long offA = (long long)zb * sAb + (long long)zh * sAh;
  const long long offB = (long long)zb * sBb + (long long)zh * sBh;
  const long long offC = (long long)zb * sCb + (long long)zh * sCh;

  const int blockRow = blockIdx.y * BM;
  const int blockCol = blockIdx.x * BN;
  const bool k8 = (K & 7) == 0;

  v8f acc[2][4] = {};

  for (int k0 = 0; k0 < K; k0 += BK) {
    __syncthreads();

    // ---- stage A tile (128x64 bf16): pure copy ----
    if (k8) {
#pragma unroll
      for (int it = 0; it < (BM * BK / 8) / 256; ++it) {   // 4 iters, 16B units
        int idx = tid + it * 256;
        int r  = idx >> 3;            // 8 octets per row
        int kq = (idx & 7) << 3;
        int gr = blockRow + r, gk = k0 + kq;
        u16x8 w = {};
        if (gr < M && gk < K)
          w = *(const u16x8*)(A + offA + (long long)gr * lda + gk);
        *(u16x8*)&As[r][kq] = w;
      }
    } else {
#pragma unroll
      for (int it = 0; it < (BM * BK / 4) / 256; ++it) {   // 8 iters, 8B units
        int idx = tid + it * 256;
        int r  = idx >> 4;
        int kq = (idx & 15) << 2;
        int gr = blockRow + r, gk = k0 + kq;
        u16x4 w = {};
        if (gr < M && gk < K)
          w = *(const u16x4*)(A + offA + (long long)gr * lda + gk);
        *(u16x4*)&As[r][kq] = w;
      }
    }

    // ---- stage B tile into Bs[n][k] (N-major) ----
    if (transB) {
      const u16* B = (const u16*)Bv;   // bf16 N-major (pre-transposed)
      if (k8) {
#pragma unroll
        for (int it = 0; it < (BN * BK / 8) / 256; ++it) { // 4 iters
          int idx = tid + it * 256;
          int n  = idx >> 3;
          int kq = (idx & 7) << 3;
          int gn = blockCol + n, gk = k0 + kq;
          u16x8 w = {};
          if (gn < N && gk < K)
            w = *(const u16x8*)(B + offB + (long long)gn * ldb + gk);
          *(u16x8*)&Bs[n][kq] = w;
        }
      } else {
#pragma unroll
        for (int it = 0; it < (BN * BK / 4) / 256; ++it) { // 8 iters
          int idx = tid + it * 256;
          int n  = idx >> 4;
          int kq = (idx & 15) << 2;
          int gn = blockCol + n, gk = k0 + kq;
          u16x4 w = {};
          if (gn < N && gk < K)
            w = *(const u16x4*)(B + offB + (long long)gn * ldb + gk);
          *(u16x4*)&Bs[n][kq] = w;
        }
      }
    } else {
      const float* B = (const float*)Bv; // f32 K-major: 4x4 register transpose
#pragma unroll
      for (int it = 0; it < ((BK / 4) * (BN / 4)) / 256; ++it) { // 2 iters
        int blk = tid + it * 256;
        int bk = blk >> 5;          // 0..15  (k block of 4)
        int bn = blk & 31;          // 0..31  (n block of 4)
        int gk0 = k0 + bk * 4, gn0 = blockCol + bn * 4;
        f32x4 r0 = {}, r1 = {}, r2 = {}, r3 = {};
        if (gn0 < N && gk0 < K) {
          const float* p = B + offB + (long long)gk0 * ldb + gn0;
          r0 = *(const f32x4*)p;
          r1 = *(const f32x4*)(p + ldb);
          r2 = *(const f32x4*)(p + 2 * ldb);
          r3 = *(const f32x4*)(p + 3 * ldb);
        }
        int kq = bk * 4, nq = bn * 4;
        u16x4 w0 = { f2bf(r0[0]), f2bf(r1[0]), f2bf(r2[0]), f2bf(r3[0]) };
        u16x4 w1 = { f2bf(r0[1]), f2bf(r1[1]), f2bf(r2[1]), f2bf(r3[1]) };
        u16x4 w2 = { f2bf(r0[2]), f2bf(r1[2]), f2bf(r2[2]), f2bf(r3[2]) };
        u16x4 w3 = { f2bf(r0[3]), f2bf(r1[3]), f2bf(r2[3]), f2bf(r3[3]) };
        *(u16x4*)&Bs[nq + 0][kq] = w0;
        *(u16x4*)&Bs[nq + 1][kq] = w1;
        *(u16x4*)&Bs[nq + 2][kq] = w2;
        *(u16x4*)&Bs[nq + 3][kq] = w3;
      }
    }

    // prefetch next A k-tile while this tile computes (global_prefetch_b8)
    if (k0 + BK < K) {
      int pr = tid & 127;
      __builtin_prefetch(A + offA + (long long)(blockRow + pr) * lda + (k0 + BK), 0, 0);
    }

    __syncthreads();

    // ---- 2 x (K=32) WMMA substeps: 16 v_wmma per iteration ----
    const int ar = lane & 15;
    const int kA = (lane >> 4) * 8;
    const int kB = (lane >> 4) * 8;
#pragma unroll
    for (int ks = 0; ks < BK; ks += 32) {
      v16bf a0 = load_frag_A(&As[wm + ar][ks],      kA);
      v16bf a1 = load_frag_A(&As[wm + 16 + ar][ks], kA);
#pragma unroll
      for (int j = 0; j < 4; ++j) {
        v16bf bf = load_frag_B(&Bs[wn + j * 16 + ar][ks], kB);
        acc[0][j] = wmma_bf16(a0, bf, acc[0][j]);
        acc[1][j] = wmma_bf16(a1, bf, acc[1][j]);
      }
    }
  }

  // Epilogue: C/D layout — lane L: col = L&15 ; row = (L>>4)*8 + vgpr_idx
  const int cc   = lane & 15;
  const int rsub = (lane >> 4) * 8;
#pragma unroll
  for (int fi = 0; fi < 2; ++fi)
#pragma unroll
    for (int fj = 0; fj < 4; ++fj)
#pragma unroll
      for (int r = 0; r < 8; ++r) {
        int row = blockRow + wm + fi * 16 + rsub + r;
        int col = blockCol + wn + fj * 16 + cc;
        if (row < M && col < N) {
          long long o = offC + (long long)row * ldc + col;
          float val = acc[fi][fj][r];
          if (bias)  val += bias[col];
          if (relu)  val = fmaxf(val, 0.f);
          if (resid) val += resid[o];
          if (Cf)    Cf[o]  = val;
          if (Cbf)   Cbf[o] = f2bf(val);
        }
      }
}

// ---- Weight transpose+convert: out[b][c][r] = bf16(in[b][r][c]) -----------
__global__ __launch_bounds__(256) void transpose_bf16_kernel(
    const float* __restrict__ in, u16* __restrict__ out, int R, int C)
{
  long long b = blockIdx.z;
  int c = blockIdx.y;
  int r = blockIdx.x * 256 + threadIdx.x;
  if (r < R)
    out[(b * C + c) * (long long)R + r] = f2bf(in[(b * R + r) * (long long)C + c]);
}

// ---- Embedding: x[b,t,:] = tok_emb[idx[b,t],:] + pos_emb[t,:] -------------
__global__ __launch_bounds__(224) void embed_kernel(
    const int* __restrict__ idx, const float* __restrict__ tok,
    const float* __restrict__ pos, float* __restrict__ x)
{
  int row = blockIdx.x, c = threadIdx.x;
  if (c < NE) {
    int t = row % TT;
    x[(long long)row * NE + c] =
        tok[(long long)idx[row] * NE + c] + pos[t * NE + c];
  }
}

// ---- LayerNorm over NE=208 (f32 in, bf16 out; one block per row) ----------
__global__ __launch_bounds__(256) void layernorm_kernel(
    const float* __restrict__ x, const float* __restrict__ g,
    const float* __restrict__ b, u16* __restrict__ out)
{
  __shared__ float red[256];
  int row = blockIdx.x, c = threadIdx.x;
  float v = (c < NE) ? x[(long long)row * NE + c] : 0.f;
  red[c] = v; __syncthreads();
  for (int s = 128; s > 0; s >>= 1) { if (c < s) red[c] += red[c + s]; __syncthreads(); }
  float mu = red[0] * (1.0f / NE); __syncthreads();
  float d = (c < NE) ? (v - mu) : 0.f;
  red[c] = d * d; __syncthreads();
  for (int s = 128; s > 0; s >>= 1) { if (c < s) red[c] += red[c + s]; __syncthreads(); }
  float inv = rsqrtf(red[0] * (1.0f / NE) + 1e-5f);
  if (c < NE) out[(long long)row * NE + c] = f2bf((v - mu) * inv * g[c] + b[c]);
}

// ---- Causal softmax: f32 scores in, bf16 probs out ------------------------
__global__ __launch_bounds__(256) void softmax_kernel(
    const float* __restrict__ s, u16* __restrict__ p, float scale)
{
  __shared__ float red[256];
  long long base = (long long)blockIdx.x * TT;
  int t = blockIdx.x % TT;          // query position
  int c = threadIdx.x;
  bool valid = (c <= t);
  float v = valid ? s[base + c] * scale : -3.0e38f;
  red[c] = v; __syncthreads();
  for (int st = 128; st > 0; st >>= 1) { if (c < st) red[c] = fmaxf(red[c], red[c + st]); __syncthreads(); }
  float mx = red[0]; __syncthreads();
  float e = valid ? __expf(v - mx) : 0.f;
  red[c] = e; __syncthreads();
  for (int st = 128; st > 0; st >>= 1) { if (c < st) red[c] += red[c + st]; __syncthreads(); }
  p[base + c] = f2bf(e * (1.0f / red[0]));
}

// ---- NLL loss over logits ([BT, VOCAB]) -----------------------------------
__global__ void zero_loss(float* loss) { *loss = 0.f; }

__global__ __launch_bounds__(256) void loss_kernel(
    const float* __restrict__ logits, const int* __restrict__ targets,
    float* __restrict__ loss)
{
  __shared__ float red[256];
  long long base = (long long)blockIdx.x * VOCAB;
  int c = threadIdx.x;
  float mx = -3.0e38f;
  for (int j = c; j < VOCAB; j += 256) mx = fmaxf(mx, logits[base + j]);
  red[c] = mx; __syncthreads();
  for (int st = 128; st > 0; st >>= 1) { if (c < st) red[c] = fmaxf(red[c], red[c + st]); __syncthreads(); }
  mx = red[0]; __syncthreads();
  float sum = 0.f;
  for (int j = c; j < VOCAB; j += 256) sum += __expf(logits[base + j] - mx);
  red[c] = sum; __syncthreads();
  for (int st = 128; st > 0; st >>= 1) { if (c < st) red[c] += red[c + st]; __syncthreads(); }
  if (c == 0) {
    float lse = mx + __logf(red[0]);
    float lt  = logits[base + targets[blockIdx.x]];
    atomicAdd(loss, (lse - lt) * (1.0f / (float)BT));
  }
}

// ---- Orchestration --------------------------------------------------------
extern "C" void kernel_launch(void* const* d_in, const int* in_sizes, int n_in,
                              void* d_out, int out_size, void* d_ws, size_t ws_size,
                              hipStream_t stream)
{
  (void)in_sizes; (void)n_in; (void)out_size; (void)ws_size;
  const int*   idx     = (const int*)  d_in[0];
  const int*   targets = (const int*)  d_in[1];
  const float* tok_emb = (const float*)d_in[2];
  const float* pos_emb = (const float*)d_in[3];
  const float* Wq   = (const float*)d_in[4];
  const float* Wk   = (const float*)d_in[5];
  const float* Wv   = (const float*)d_in[6];
  const float* Wo   = (const float*)d_in[7];
  const float* bo   = (const float*)d_in[8];
  const float* ln1g = (const float*)d_in[9];
  const float* ln1b = (const float*)d_in[10];
  const float* W1   = (const float*)d_in[11];
  const float* b1   = (const float*)d_in[12];
  const float* W2   = (const float*)d_in[13];
  const float* b2   = (const float*)d_in[14];
  const float* ln2g = (const float*)d_in[15];
  const float* ln2b = (const float*)d_in[16];
  const float* lnfg = (const float*)d_in[17];
  const float* lnfb = (const float*)d_in[18];
  const float* Wlm  = (const float*)d_in[19];
  const float* blm  = (const float*)d_in[20];

  float* logits = (float*)d_out;
  float* loss   = logits + (long long)BT * VOCAB;

  // ---- workspace carve-out ----
  size_t off = 0;
  auto take = [&](long long bytes) -> void* {
    void* p = (char*)d_ws + off;
    off = (off + (size_t)bytes + 255) & ~(size_t)255;
    return p;
  };
  float* x      = (float*)take((long long)BT * NE * 4);
  float* v      = (float*)take((long long)BT * NE * 4);
  float* scores = (float*)take((long long)BB * NHD * TT * TT * 4);
  u16*   hb     = (u16*)  take((long long)BT * NE * 2);
  u16*   qb     = (u16*)  take((long long)BT * NE * 2);   // also attn-out
  u16*   kbuf   = (u16*)  take((long long)BT * NE * 2);
  u16*   bigb   = (u16*)  take((long long)BB * NHD * TT * TT * 2); // probs / mlp-hidden
  u16*   WqT    = (u16*)  take((long long)NL * NHD * HD * NE * 2);
  u16*   WkT    = (u16*)  take((long long)NL * NHD * HD * NE * 2);
  u16*   WvT    = (u16*)  take((long long)NL * NHD * HD * NE * 2);
  u16*   WoT    = (u16*)  take((long long)NL * NE * NE * 2);
  u16*   W1T    = (u16*)  take((long long)NL * FF * NE * 2);
  u16*   W2T    = (u16*)  take((long long)NL * NE * FF * 2);
  u16*   WlmT   = (u16*)  take((long long)VOCAB * NE * 2);

  // ---- one-time weight transpose+convert to bf16 N-major ----
  auto tconv = [&](const float* in, u16* out, int R, int C, int nb) {
    dim3 grid((R + 255) / 256, C, nb);
    transpose_bf16_kernel<<<grid, 256, 0, stream>>>(in, out, R, C);
  };
  tconv(Wq,  WqT,  NE, HD, NL * NHD);
  tconv(Wk,  WkT,  NE, HD, NL * NHD);
  tconv(Wv,  WvT,  NE, HD, NL * NHD);
  tconv(Wo,  WoT,  NE, NE, NL);
  tconv(W1,  W1T,  NE, FF, NL);
  tconv(W2,  W2T,  FF, NE, NL);
  tconv(Wlm, WlmT, NE, VOCAB, 1);

  auto gemm = [&](const u16* A, const void* Bp, const float* bias,
                  const float* res, float* Cf, u16* Cbf, int M, int N, int K,
                  int lda, int ldb, int ldc,
                  long long sAb, long long sAh, long long sBb, long long sBh,
                  long long sCb, long long sCh, int nb, int nh, int tb, int rl) {
    dim3 grid((N + BN - 1) / BN, (M + BM - 1) / BM, nb);
    gemm_bf16_kernel<<<grid, 256, 0, stream>>>(A, Bp, bias, res, Cf, Cbf,
        M, N, K, lda, ldb, ldc, sAb, sAh, sBb, sBh, sCb, sCh, nh, tb, rl);
  };

  embed_kernel<<<BT, 224, 0, stream>>>(idx, tok_emb, pos_emb, x);
  const float scale = 1.0f / sqrtf((float)NE);   // reference scales by C**-0.5

  const long long whs = (long long)HD * NE;      // transposed per-head weight stride
  for (int L = 0; L < NL; ++L) {
    layernorm_kernel<<<BT, 256, 0, stream>>>(x, ln1g + L * NE, ln1b + L * NE, hb);

    // Q/K/V: per-head [BT,208]x[208,52] batched over heads, packed at col h*52
    gemm(hb, WqT + (long long)L * NHD * whs, nullptr, nullptr, nullptr, qb,
         BT, HD, NE, NE, NE, NE, 0, 0, 0, whs, 0, (long long)HD, NHD, NHD, 1, 0);
    gemm(hb, WkT + (long long)L * NHD * whs, nullptr, nullptr, nullptr, kbuf,
         BT, HD, NE, NE, NE, NE, 0, 0, 0, whs, 0, (long long)HD, NHD, NHD, 1, 0);
    gemm(hb, WvT + (long long)L * NHD * whs, nullptr, nullptr, v, nullptr,
         BT, HD, NE, NE, NE, NE, 0, 0, 0, whs, 0, (long long)HD, NHD, NHD, 1, 0);

    // scores[b,h] = Q Kᵀ : [T,52]x[52,T], batched over 1024 (b,h)
    gemm(qb, kbuf, nullptr, nullptr, scores, nullptr, TT, TT, HD, NE, NE, TT,
         (long long)TT * NE, (long long)HD, (long long)TT * NE, (long long)HD,
         (long long)NHD * TT * TT, (long long)TT * TT, BB * NHD, NHD, 1, 0);

    softmax_kernel<<<BB * NHD * TT, TT, 0, stream>>>(scores, bigb, scale);

    // o[b,h] = P V : [T,T](bf16) x [T,52](f32, K-major path) -> qb (bf16)
    gemm(bigb, (const void*)v, nullptr, nullptr, nullptr, qb, TT, HD, TT,
         TT, NE, NE,
         (long long)NHD * TT * TT, (long long)TT * TT,
         (long long)TT * NE, (long long)HD,
         (long long)TT * NE, (long long)HD, BB * NHD, NHD, 0, 0);

    // x = x + o @ Wo + bo
    gemm(qb, WoT + (long long)L * NE * NE, bo + L * NE, x, x, nullptr,
         BT, NE, NE, NE, NE, NE, 0, 0, 0, 0, 0, 0, 1, 1, 1, 0);

    layernorm_kernel<<<BT, 256, 0, stream>>>(x, ln2g + L * NE, ln2b + L * NE, hb);

    // hid = relu(h @ W1 + b1) (bf16) ; x = x + hid @ W2 + b2
    gemm(hb, W1T + (long long)L * FF * NE, b1 + (long long)L * FF, nullptr,
         nullptr, bigb, BT, FF, NE, NE, NE, FF, 0, 0, 0, 0, 0, 0, 1, 1, 1, 1);
    gemm(bigb, W2T + (long long)L * NE * FF, b2 + L * NE, x, x, nullptr,
         BT, NE, FF, FF, FF, NE, 0, 0, 0, 0, 0, 0, 1, 1, 1, 0);
  }

  layernorm_kernel<<<BT, 256, 0, stream>>>(x, lnfg, lnfb, hb);
  gemm(hb, WlmT, blm, nullptr, logits, nullptr, BT, VOCAB, NE, NE, NE, VOCAB,
       0, 0, 0, 0, 0, 0, 1, 1, 1, 0);

  zero_loss<<<1, 1, 0, stream>>>(loss);
  loss_kernel<<<BT, 256, 0, stream>>>(logits, targets, loss);
}
